// MultiHeadedAttention_46651934769612
// MI455X (gfx1250) — matrix operand order repaired and checked by reading, not compiled, not run on hardware
//
#include <hip/hip_runtime.h>
#include <hip/hip_bf16.h>

// ---------------------------------------------------------------------------
// MultiHeadedAttention for MI455X (gfx1250): bf16 WMMA everywhere.
// B=4, S=2048, H=1024, NH=16, DK=64
// Pipeline:
//   convert W* -> bf16 (once), convert X -> bf16 (per projection, buffer reused)
//   gemm_bf16 (4x4 register-blocked WMMA)   -> Q/K/V (B,NH,S,DK) bf16
//   flash_attn (online softmax, WMMA QK^T and PV) -> O (B,S,H) bf16
//   gemm_bf16 dense epilogue (+bias, fp32)  -> out
// ---------------------------------------------------------------------------

#define B_SZ   4
#define S_LEN  2048
#define H_DIM  1024
#define NH_C   16
#define DK_C   64

typedef __attribute__((ext_vector_type(16))) __bf16 v16bf;
typedef __attribute__((ext_vector_type(8)))  float  v8f;

union FragB16 { uint4 q[2]; v16bf v; };
union FragF32 { v8f v; float f[8]; };

__device__ __forceinline__ unsigned short f2bf(float x) {
  unsigned u = __float_as_uint(x);
  unsigned r = u + 0x7FFFu + ((u >> 16) & 1u);   // round-to-nearest-even
  return (unsigned short)(r >> 16);
}
__device__ __forceinline__ unsigned pack2(float lo, float hi) {
  return (unsigned)f2bf(lo) | ((unsigned)f2bf(hi) << 16);
}

// ---------------------------------------------------------------------------
// fp32 -> bf16 converter: each thread handles 8 elements (2x b128 in, 1x b128 out)
// ---------------------------------------------------------------------------
__global__ __launch_bounds__(256)
void cvt_bf16_kernel(const float* __restrict__ src, unsigned short* __restrict__ dst,
                     long long n8)  // n/8
{
  long long t = (long long)blockIdx.x * blockDim.x + threadIdx.x;
  if (t >= n8) return;
  const float4 a = *(const float4*)(src + t * 8);
  const float4 b = *(const float4*)(src + t * 8 + 4);
  uint4 o;
  o.x = pack2(a.x, a.y); o.y = pack2(a.z, a.w);
  o.z = pack2(b.x, b.y); o.w = pack2(b.z, b.w);
  *(uint4*)(dst + t * 8) = o;
}

// ---------------------------------------------------------------------------
// NT GEMM, all-bf16 operands: C[m,n] = sum_k A[m,k] * W[n,k]
// Block = 256 threads = 8 waves arranged 2(M) x 4(N); each wave computes a
// 64x64 tile as 4x4 WMMA tiles => per K-step: 8 frag loads feed 16 wmma.
// Block tile 128(M) x 256(N).
// ---------------------------------------------------------------------------
template<bool SPLIT_HEADS>
__global__ __launch_bounds__(256)
void gemm_bf16_kernel(const unsigned short* __restrict__ A,
                      const unsigned short* __restrict__ Wb,
                      const float* __restrict__ bias,
                      void* __restrict__ Outp,
                      int M, int N, int K)
{
  const int lane = threadIdx.x & 31;
  const int wave = threadIdx.x >> 5;
  const int wm   = wave >> 2;          // 0..1
  const int wn   = wave & 3;           // 0..3
  const int lm   = lane & 15;
  const int hi   = lane >> 4;          // lane half

  const int row0 = blockIdx.x * 128 + wm * 64;
  const int n0   = blockIdx.y * 256 + wn * 64;

  const unsigned short* aBase[4];
  const unsigned short* bBase[4];
#pragma unroll
  for (int i = 0; i < 4; ++i)
    aBase[i] = A  + (size_t)(row0 + i * 16 + lm) * K + hi * 8;
#pragma unroll
  for (int j = 0; j < 4; ++j)
    bBase[j] = Wb + (size_t)(n0 + j * 16 + lm) * K + hi * 8;

  FragF32 acc[4][4];
#pragma unroll
  for (int i = 0; i < 4; ++i)
#pragma unroll
    for (int j = 0; j < 4; ++j)
#pragma unroll
      for (int r = 0; r < 8; ++r) acc[i][j].f[r] = 0.0f;

  for (int kb = 0; kb < K; kb += 32) {
    __builtin_prefetch(aBase[0] + kb + 32, 0, 0);
    __builtin_prefetch(bBase[0] + kb + 32, 0, 0);
    FragB16 af[4], bf[4];
#pragma unroll
    for (int i = 0; i < 4; ++i) {
      af[i].q[0] = *(const uint4*)(aBase[i] + kb);
      af[i].q[1] = *(const uint4*)(aBase[i] + kb + 16);
    }
#pragma unroll
    for (int j = 0; j < 4; ++j) {
      bf[j].q[0] = *(const uint4*)(bBase[j] + kb);
      bf[j].q[1] = *(const uint4*)(bBase[j] + kb + 16);
    }
#pragma unroll
    for (int i = 0; i < 4; ++i)
#pragma unroll
      for (int j = 0; j < 4; ++j)
        acc[i][j].v = __builtin_amdgcn_wmma_f32_16x16x32_bf16(
            false, af[i].v, false, bf[j].v, (short)0, acc[i][j].v, false, false);
  }

  // ---- epilogue
#pragma unroll
  for (int i = 0; i < 4; ++i)
#pragma unroll
    for (int j = 0; j < 4; ++j)
#pragma unroll
      for (int r = 0; r < 8; ++r) {
        const int row = row0 + i * 16 + r + hi * 8;
        const int col = n0 + j * 16 + lm;
        const float cv = acc[i][j].f[r];
        if (SPLIT_HEADS) {
          const int bi = row >> 11;            // / S_LEN
          const int si = row & (S_LEN - 1);
          const int h  = col >> 6;             // / DK
          const int d  = col & (DK_C - 1);
          ((unsigned short*)Outp)[(((size_t)(bi * NH_C + h)) * S_LEN + si) * DK_C + d] = f2bf(cv);
        } else {
          ((float*)Outp)[(size_t)row * N + col] = cv + bias[col];
        }
      }
}

// ---------------------------------------------------------------------------
// Flash attention: one block per (b*NH+h, q-tile of 128 rows); 8 waves,
// each wave owns 16 query rows. Streams 32-key tiles through LDS with
// online softmax; QK^T and PV both use v_wmma_f32_16x16x32_bf16.
// ---------------------------------------------------------------------------
__global__ __launch_bounds__(256)
void flash_attn_kernel(const unsigned short* __restrict__ Qb,
                       const unsigned short* __restrict__ Kb,
                       const unsigned short* __restrict__ Vb,
                       const int* __restrict__ mask,
                       unsigned short* __restrict__ Ob)
{
  __shared__ unsigned short Kt[32 * 64];      // [key][d]      (4 KB)
  __shared__ unsigned short Vt[64 * 32];      // [d][key]      (4 KB)
  __shared__ unsigned short Ps[8][16 * 32];   // per-wave P    (8 KB)

  const int bh   = blockIdx.x;                // b*NH + h
  const int bi   = bh >> 4;
  const int h    = bh & 15;
  const int lane = threadIdx.x & 31;
  const int wave = threadIdx.x >> 5;
  const int lm   = lane & 15;
  const int hi   = lane >> 4;
  const int q0   = blockIdx.y * 128 + wave * 16;

  const unsigned short* Qbase = Qb + (size_t)bh * S_LEN * DK_C;
  const unsigned short* Kbase = Kb + (size_t)bh * S_LEN * DK_C;
  const unsigned short* Vbase = Vb + (size_t)bh * S_LEN * DK_C;

  // Q fragments for kdim 0..31 and 32..63, resident across the whole K loop
  FragB16 qf[2];
  {
    const unsigned short* qp = Qbase + (size_t)(q0 + lm) * DK_C + hi * 8;
    qf[0].q[0] = *(const uint4*)(qp);
    qf[0].q[1] = *(const uint4*)(qp + 16);
    qf[1].q[0] = *(const uint4*)(qp + 32);
    qf[1].q[1] = *(const uint4*)(qp + 48);
  }

  float rowbias[8];
#pragma unroll
  for (int r = 0; r < 8; ++r)
    rowbias[r] = mask[bi * S_LEN + q0 + r + hi * 8] ? 0.0f : -1e9f;

  float mrow[8], lrow[8];
  FragF32 accd[4];
#pragma unroll
  for (int r = 0; r < 8; ++r) {
    mrow[r] = -1e30f; lrow[r] = 0.0f;
    accd[0].f[r] = accd[1].f[r] = accd[2].f[r] = accd[3].f[r] = 0.0f;
  }

  for (int kt = 0; kt < S_LEN / 32; ++kt) {
    const int kbase = kt * 32;
    __syncthreads();
    // ---- stage K tile (row-major) and V tile (transposed) into LDS
    {
      const int t    = threadIdx.x;
      const int krow = t >> 3;          // 0..31
      const int kcol = (t & 7) * 8;     // 0..56
      *(uint4*)&Kt[krow * 64 + kcol] =
          *(const uint4*)(Kbase + (size_t)(kbase + krow) * DK_C + kcol);
      union { uint4 q; unsigned short s[8]; } vu;
      vu.q = *(const uint4*)(Vbase + (size_t)(kbase + krow) * DK_C + kcol);
#pragma unroll
      for (int j = 0; j < 8; ++j)
        Vt[(kcol + j) * 32 + krow] = vu.s[j];
    }
    __syncthreads();

    // ---- scores: 16 queries x 32 keys (2 N-halves x 2 kdim steps)
    FragF32 s[2];
#pragma unroll
    for (int r = 0; r < 8; ++r) { s[0].f[r] = 0.0f; s[1].f[r] = 0.0f; }
#pragma unroll
    for (int nf = 0; nf < 2; ++nf) {
#pragma unroll
      for (int kb = 0; kb < 2; ++kb) {
        FragB16 kf;
        const unsigned short* kp = &Kt[(nf * 16 + lm) * 64 + kb * 32 + hi * 8];
        kf.q[0] = *(const uint4*)(kp);
        kf.q[1] = *(const uint4*)(kp + 16);
        s[nf].v = __builtin_amdgcn_wmma_f32_16x16x32_bf16(false, qf[kb].v, false, kf.v,
                                                          (short)0, s[nf].v, false, false);
      }
    }

    // ---- scale + mask + online softmax
    const float cb0 = mask[bi * S_LEN + kbase + lm]      ? 0.0f : -1e9f;
    const float cb1 = mask[bi * S_LEN + kbase + 16 + lm] ? 0.0f : -1e9f;
    float tscale[8];
#pragma unroll
    for (int r = 0; r < 8; ++r) {
      float s0 = s[0].f[r] * 0.125f + cb0 + rowbias[r];
      float s1 = s[1].f[r] * 0.125f + cb1 + rowbias[r];
      float tm = fmaxf(s0, s1);
#pragma unroll
      for (int xm = 1; xm <= 8; xm <<= 1)           // stays in 16-lane half
        tm = fmaxf(tm, __shfl_xor(tm, xm, 32));
      const float mnew = fmaxf(mrow[r], tm);
      const float sc   = __expf(mrow[r] - mnew);
      const float p0   = __expf(s0 - mnew);
      const float p1   = __expf(s1 - mnew);
      s[0].f[r] = p0; s[1].f[r] = p1;
      float rs = p0 + p1;
#pragma unroll
      for (int xm = 1; xm <= 8; xm <<= 1)
        rs += __shfl_xor(rs, xm, 32);
      lrow[r]   = lrow[r] * sc + rs;
      mrow[r]   = mnew;
      tscale[r] = sc;
    }
#pragma unroll
    for (int c = 0; c < 4; ++c)
#pragma unroll
      for (int r = 0; r < 8; ++r)
        accd[c].f[r] *= tscale[r];

    // ---- P: C-layout -> A-frag layout via per-wave LDS round trip
    unsigned short* pw = Ps[wave];
#pragma unroll
    for (int nf = 0; nf < 2; ++nf)
#pragma unroll
      for (int r = 0; r < 8; ++r)
        pw[(r + hi * 8) * 32 + nf * 16 + lm] = f2bf(s[nf].f[r]);
    FragB16 pf;
    {
      const unsigned short* pp = pw + lm * 32 + hi * 8;
      pf.q[0] = *(const uint4*)(pp);
      pf.q[1] = *(const uint4*)(pp + 16);
    }

    // ---- PV: accumulate 16 x 64 output in f32
#pragma unroll
    for (int c = 0; c < 4; ++c) {
      FragB16 vf;
      const unsigned short* vp = &Vt[(c * 16 + lm) * 32 + hi * 8];
      vf.q[0] = *(const uint4*)(vp);
      vf.q[1] = *(const uint4*)(vp + 16);
      accd[c].v = __builtin_amdgcn_wmma_f32_16x16x32_bf16(false, pf.v, false, vf.v,
                                                          (short)0, accd[c].v, false, false);
    }
  }

  // ---- normalize and store O in (B, S, H) bf16 layout
#pragma unroll
  for (int c = 0; c < 4; ++c)
#pragma unroll
    for (int r = 0; r < 8; ++r) {
      const int row = q0 + r + hi * 8;
      const int col = h * DK_C + c * 16 + lm;
      const float ov = accd[c].f[r] / lrow[r];
      Ob[(size_t)(bi * S_LEN + row) * H_DIM + col] = f2bf(ov);
    }
}

// ---------------------------------------------------------------------------
extern "C" void kernel_launch(void* const* d_in, const int* in_sizes, int n_in,
                              void* d_out, int out_size, void* d_ws, size_t ws_size,
                              hipStream_t stream) {
  const float* query = (const float*)d_in[0];
  const float* key_t = (const float*)d_in[1];
  const float* value = (const float*)d_in[2];
  const int*   mask  = (const int*)d_in[3];
  const float* Wq    = (const float*)d_in[4];
  const float* Wk    = (const float*)d_in[5];
  const float* Wv    = (const float*)d_in[6];
  const float* Wo    = (const float*)d_in[7];
  const float* bo    = (const float*)d_in[8];

  const int M = B_SZ * S_LEN;   // 8192
  const int N = H_DIM;          // 1024
  const int K = H_DIM;          // 1024

  // workspace layout (bf16 elements): Qb,Kb,Vb,Ob (M*N each), Xb (M*K),
  // 4 weight copies (N*K each)  -> 88 MB total
  unsigned short* Qb  = (unsigned short*)d_ws;
  unsigned short* Kb  = Qb  + (size_t)M * N;
  unsigned short* Vb  = Kb  + (size_t)M * N;
  unsigned short* Ob  = Vb  + (size_t)M * N;
  unsigned short* Xb  = Ob  + (size_t)M * N;
  unsigned short* Wqb = Xb  + (size_t)M * K;
  unsigned short* Wkb = Wqb + (size_t)N * K;
  unsigned short* Wvb = Wkb + (size_t)N * K;
  unsigned short* Wob = Wvb + (size_t)N * K;

  const long long nX8 = (long long)M * K / 8;       // 1M threads
  const long long nW8 = (long long)N * K / 8;       // 128K threads
  const int CT = 256;

  // convert weights once
  cvt_bf16_kernel<<<dim3((unsigned)((nW8 + CT - 1) / CT)), CT, 0, stream>>>(Wq, Wqb, nW8);
  cvt_bf16_kernel<<<dim3((unsigned)((nW8 + CT - 1) / CT)), CT, 0, stream>>>(Wk, Wkb, nW8);
  cvt_bf16_kernel<<<dim3((unsigned)((nW8 + CT - 1) / CT)), CT, 0, stream>>>(Wv, Wvb, nW8);
  cvt_bf16_kernel<<<dim3((unsigned)((nW8 + CT - 1) / CT)), CT, 0, stream>>>(Wo, Wob, nW8);

  dim3 gg(M / 128, N / 256);    // (64, 4)
  dim3 bb(256);

  // Q projection
  cvt_bf16_kernel<<<dim3((unsigned)((nX8 + CT - 1) / CT)), CT, 0, stream>>>(query, Xb, nX8);
  gemm_bf16_kernel<true ><<<gg, bb, 0, stream>>>(Xb, Wqb, nullptr, Qb, M, N, K);
  // K projection (Xb reused; stream-ordered)
  cvt_bf16_kernel<<<dim3((unsigned)((nX8 + CT - 1) / CT)), CT, 0, stream>>>(key_t, Xb, nX8);
  gemm_bf16_kernel<true ><<<gg, bb, 0, stream>>>(Xb, Wkb, nullptr, Kb, M, N, K);
  // V projection
  cvt_bf16_kernel<<<dim3((unsigned)((nX8 + CT - 1) / CT)), CT, 0, stream>>>(value, Xb, nX8);
  gemm_bf16_kernel<true ><<<gg, bb, 0, stream>>>(Xb, Wvb, nullptr, Vb, M, N, K);

  // attention
  dim3 ga(B_SZ * NH_C, S_LEN / 128);   // (64, 16)
  flash_attn_kernel<<<ga, dim3(256), 0, stream>>>(Qb, Kb, Vb, mask, Ob);

  // output projection (+bias, fp32 out)
  gemm_bf16_kernel<false><<<gg, bb, 0, stream>>>(Ob, Wob, bo, (float*)d_out, M, N, K);
}
